// GlobalCoherenceLayer_21431886807334
// MI455X (gfx1250) — matrix operand order repaired and checked by reading, not compiled, not run on hardware
//
#include <hip/hip_runtime.h>
#include <hip/hip_bf16.h>
#include <math.h>

#define B_ 2
#define T_ 2048
#define DM_ 1024
#define H_ 16
#define D_ 64
#define TOPK_ 204
#define NEG_ (-1e9f)
#define SCALE_ 8.0f            // sqrt(D)
#define LN_EPS_ 1e-5f

typedef __attribute__((ext_vector_type(16))) unsigned short v16u16;
typedef __attribute__((ext_vector_type(8)))  unsigned short v8u16;
typedef __attribute__((ext_vector_type(16))) __bf16         v16bf;
typedef __attribute__((ext_vector_type(8)))  float          v8f;

// ---------- helpers ----------
__device__ inline unsigned short f2bf(float f) {       // fp32 -> bf16 RNE
  unsigned int u = __float_as_uint(f);
  unsigned int r = u + 0x7FFFu + ((u >> 16) & 1u);
  return (unsigned short)(r >> 16);
}

__device__ inline v8f wmma_bf16(v16u16 a, v16u16 b, v8f c) {
  return __builtin_amdgcn_wmma_f32_16x16x32_bf16(
      false, __builtin_bit_cast(v16bf, a),
      false, __builtin_bit_cast(v16bf, b),
      (short)0, c, false, false);
}

__device__ inline float wave_sum(float v) {
  #pragma unroll
  for (int off = 16; off > 0; off >>= 1) v += __shfl_xor(v, off, 32);
  return v;
}
__device__ inline float wave_max(float v) {
  #pragma unroll
  for (int off = 16; off > 0; off >>= 1) v = fmaxf(v, __shfl_xor(v, off, 32));
  return v;
}
__device__ inline float wave_min(float v) {
  #pragma unroll
  for (int off = 16; off > 0; off >>= 1) v = fminf(v, __shfl_xor(v, off, 32));
  return v;
}
__device__ inline int wave_isum(int v) {
  #pragma unroll
  for (int off = 16; off > 0; off >>= 1) v += __shfl_xor(v, off, 32);
  return v;
}

// A-fragment (16x32 bf16): lane = {hi,row}; VGPR0-3 hold
// K = 16*hi..16*hi+7, VGPR4-7 hold K = 16*hi+8..16*hi+15 (matches b_frag).
// Two 16-byte vector loads + concat -- no per-element shuffling.
__device__ inline v16u16 a_frag(const unsigned short* row, int kk, int hi) {
  const v8u16 p0 = *(const v8u16*)(row + kk + 16 * hi);
  const v8u16 p1 = *(const v8u16*)(row + kk + 16 * hi + 8);
  return __builtin_shufflevector(p0, p1, 0,1,2,3,4,5,6,7,8,9,10,11,12,13,14,15);
}
// B-fragment (32x16 bf16): lane = col, 16 contiguous K values -> one 32B load.
__device__ inline v16u16 b_frag(const unsigned short* strip) {
  return *(const v16u16*)strip;
}

// ---------- kernel 0a: fp32 -> bf16 bulk convert (for x) ----------
__global__ void cvt_bf16_kernel(const float* __restrict__ src,
                                unsigned short* __restrict__ dst, int n) {
  const int i = (blockIdx.x * blockDim.x + threadIdx.x) * 8;
  if (i >= n) return;
  v8u16 p;
  #pragma unroll
  for (int j = 0; j < 8; ++j) p[j] = f2bf(src[i + j]);
  *(v8u16*)(dst + i) = p;
}

// ---------- kernel 0b: W[k][n] -> bf16 W^T[n][k] (LDS-tiled) ----------
// blockIdx.z selects which weight; 32x32 tiles, block (32,8).
__global__ void transpose_w_kernel(const float* __restrict__ Wq,
                                   const float* __restrict__ Wk,
                                   const float* __restrict__ Wv,
                                   const float* __restrict__ Wo,
                                   unsigned short* __restrict__ WqT,
                                   unsigned short* __restrict__ WkT,
                                   unsigned short* __restrict__ WvT,
                                   unsigned short* __restrict__ WoT) {
  __shared__ unsigned short tile[32][33];
  const int z = blockIdx.z;
  const float* src = (z == 0) ? Wq : (z == 1) ? Wk : (z == 2) ? Wv : Wo;
  unsigned short* dst = (z == 0) ? WqT : (z == 1) ? WkT : (z == 2) ? WvT : WoT;
  const int tx = threadIdx.x, ty = threadIdx.y;
  const int n0 = blockIdx.x * 32, k0 = blockIdx.y * 32;
  for (int r = ty; r < 32; r += 8)
    tile[r][tx] = f2bf(src[(size_t)(k0 + r) * DM_ + n0 + tx]);
  __syncthreads();
  for (int r = ty; r < 32; r += 8)
    dst[(size_t)(n0 + r) * DM_ + k0 + tx] = tile[tx][r];
}

// ---------- kernel 1: fused QKV projection (xbf @ {Wq,Wk,Wv}^T) ----------
// One wave -> one 16x16 tile; inner loop = 2x16B A loads + 32B B load + WMMA.
// Q,K stored [B,H,T,D]; V stored transposed [B,H,D,T] for contiguous PV B-frags.
__global__ void qkv_proj_kernel(const unsigned short* __restrict__ xbf,
                                const unsigned short* __restrict__ WqT,
                                const unsigned short* __restrict__ WkT,
                                const unsigned short* __restrict__ WvT,
                                unsigned short* __restrict__ Qbf,
                                unsigned short* __restrict__ Kbf,
                                unsigned short* __restrict__ Vt) {
  const int lane = threadIdx.x & 31;
  const int wid  = blockIdx.x * (blockDim.x >> 5) + (threadIdx.x >> 5);
  const int tilesPerMat = (B_ * T_ / 16) * (DM_ / 16);   // 16384
  const int mat = wid / tilesPerMat;
  const int rem = wid % tilesPerMat;
  const int tm = rem / (DM_ / 16);
  const int tn = rem % (DM_ / 16);

  const unsigned short* WT = (mat == 0) ? WqT : (mat == 1) ? WkT : WvT;

  const int hi = lane >> 4, lo = lane & 15;
  const unsigned short* xrow = xbf + (size_t)(tm * 16 + lo) * DM_;
  const int ncol = tn * 16 + lo;
  const unsigned short* wcol = WT + (size_t)ncol * DM_;

  v8f c = {};
  for (int kk = 0; kk < DM_; kk += 32) {
    c = wmma_bf16(a_frag(xrow, kk, hi), b_frag(wcol + kk + 16 * hi), c);
  }

  const int h = ncol >> 6, d = ncol & 63;
  #pragma unroll
  for (int r = 0; r < 8; ++r) {
    const int row = tm * 16 + r + 8 * hi;
    const int bb = row >> 11, t = row & (T_ - 1);
    const unsigned short v = f2bf(c[r]);
    if (mat == 2)
      Vt[((size_t)(bb * H_ + h) * D_ + d) * T_ + t] = v;          // [B,H,D,T]
    else if (mat == 0)
      Qbf[((size_t)(bb * H_ + h) * T_ + t) * D_ + d] = v;         // [B,H,T,D]
    else
      Kbf[((size_t)(bb * H_ + h) * T_ + t) * D_ + d] = v;
  }
}

// ---------- kernel 2: attention (scores -> topk -> softmax -> PV) ----------
// One workgroup (8 waves) per (b, h, 16-query tile). Score strip 16xT (fp32,
// 128 KB) + bf16 P strip (64 KB) live in LDS -- scores never touch HBM.
__global__ void attention_kernel(const unsigned short* __restrict__ Qbf,
                                 const unsigned short* __restrict__ Kbf,
                                 const unsigned short* __restrict__ Vt,
                                 unsigned short* __restrict__ Obf,
                                 float* __restrict__ acc) {
  extern __shared__ char smem_raw[];
  float* S            = (float*)smem_raw;                              // [16][T_] fp32
  unsigned short* Pbf = (unsigned short*)(smem_raw + (size_t)16 * T_ * 4);  // [16][T_] bf16
  float* red          = (float*)(smem_raw + (size_t)16 * T_ * 4 + (size_t)16 * T_ * 2);

  const int tid  = threadIdx.x;
  const int lane = tid & 31;
  const int w    = tid >> 5;
  const int hi   = lane >> 4, lo = lane & 15;

  const int blk = blockIdx.x;
  const int qt  = blk & (T_ / 16 - 1);
  const int bh  = blk / (T_ / 16);
  const int h   = bh & (H_ - 1);
  const int bb  = bh / H_;
  const int qbase = qt * 16;

  const unsigned short* Qh  = Qbf + (size_t)(bb * H_ + h) * T_ * D_;
  const unsigned short* Kh  = Kbf + (size_t)(bb * H_ + h) * T_ * D_;
  const unsigned short* Vth = Vt  + (size_t)(bb * H_ + h) * D_ * T_;

  // --- Phase A: S = Q K^T / sqrt(D), causal mask, raw-energy accumulation ---
  const unsigned short* qrow = Qh + (size_t)(qbase + lo) * D_;
  const v16u16 aq0 = a_frag(qrow, 0, hi);
  const v16u16 aq1 = a_frag(qrow, 32, hi);

  float rawsum = 0.f;
  const float invScale = 1.0f / SCALE_;
  for (int nt = w; nt < T_ / 16; nt += 8) {
    const int tkey = nt * 16 + lo;
    const unsigned short* krow = Kh + (size_t)tkey * D_;
    v8f c = {};
    c = wmma_bf16(aq0, b_frag(krow + 16 * hi), c);
    c = wmma_bf16(aq1, b_frag(krow + 32 + 16 * hi), c);
    #pragma unroll
    for (int r = 0; r < 8; ++r) {
      const int rl = r + 8 * hi;
      const int q  = qbase + rl;
      const float val = c[r] * invScale;
      const bool causal = (tkey <= q);
      rawsum += causal ? val : 0.f;
      S[rl * T_ + tkey] = causal ? val : NEG_;
    }
  }
  rawsum = wave_sum(rawsum);
  if (lane == 0) red[w] = rawsum;
  __syncthreads();
  if (tid == 0) {
    float t = 0.f;
    #pragma unroll
    for (int i = 0; i < 8; ++i) t += red[i];
    atomicAdd(acc, t);
  }
  __syncthreads();

  // --- Phase B: per-row top-k threshold + softmax (one wave per row) ---
  for (int rl = w; rl < 16; rl += 8) {
    float* Srow = S + rl * T_;
    unsigned short* Prow = Pbf + rl * T_;
    const int q = qbase + rl;
    const int valid = q + 1;                 // causal row length

    float m = NEG_;
    for (int i = lane; i < T_; i += 32) m = fmaxf(m, Srow[i]);
    m = wave_max(m);

    if (valid > TOPK_) {
      float mn = m;
      for (int i = lane; i < valid; i += 32) mn = fminf(mn, Srow[i]);
      mn = wave_min(mn);
      float lo_ = mn, hi_ = m;
      #pragma unroll 1
      for (int it = 0; it < 22; ++it) {      // bisection on value: count(>=mid)
        const float mid = 0.5f * (lo_ + hi_);
        int cnt = 0;
        for (int i = lane; i < valid; i += 32) cnt += (Srow[i] >= mid) ? 1 : 0;
        cnt = wave_isum(cnt);
        if (cnt >= TOPK_) lo_ = mid; else hi_ = mid;
      }
      const float thr = lo_;                 // ~ k-th largest value
      for (int i = lane; i < valid; i += 32)
        if (Srow[i] < thr) Srow[i] = NEG_;
    }

    float s = 0.f;
    for (int i = lane; i < T_; i += 32) s += __expf(Srow[i] - m);
    s = wave_sum(s);
    const float inv = 1.0f / s;
    for (int i = lane; i < T_; i += 32)
      Prow[i] = f2bf(__expf(Srow[i] - m) * inv);   // bf16 P for WMMA phase
  }
  __syncthreads();

  // --- Phase C: O = P V (waves 0..3, one 16-wide d-tile each) ---
  // A-frags: ds_load_b128 pairs from Pbf; B-frags: contiguous 32B from V^T.
  if (w < 4) {
    const int dtile = w * 16;
    const int d = dtile + lo;
    const unsigned short* prow = Pbf + lo * T_;
    const unsigned short* vrow = Vth + (size_t)d * T_;
    v8f c = {};
    for (int kk = 0; kk < T_; kk += 32) {
      c = wmma_bf16(a_frag(prow, kk, hi), b_frag(vrow + kk + 16 * hi), c);
    }
    #pragma unroll
    for (int r = 0; r < 8; ++r) {
      const int q   = qbase + r + 8 * hi;
      const int col = h * D_ + dtile + lo;
      Obf[((size_t)bb * T_ + q) * DM_ + col] = f2bf(c[r]);
    }
  }
}

// ---------- kernel 3: output projection (O @ Wo^T-majored) -> fp32 ----------
__global__ void out_proj_kernel(const unsigned short* __restrict__ Obf,
                                const unsigned short* __restrict__ WoT,
                                float* __restrict__ attnOut) {
  const int lane = threadIdx.x & 31;
  const int wid  = blockIdx.x * (blockDim.x >> 5) + (threadIdx.x >> 5);
  const int tm = wid / (DM_ / 16);
  const int tn = wid % (DM_ / 16);
  const int hi = lane >> 4, lo = lane & 15;

  const unsigned short* arow = Obf + (size_t)(tm * 16 + lo) * DM_;
  const int ncol = tn * 16 + lo;
  const unsigned short* wcol = WoT + (size_t)ncol * DM_;

  v8f c = {};
  for (int kk = 0; kk < DM_; kk += 32) {
    c = wmma_bf16(a_frag(arow, kk, hi), b_frag(wcol + kk + 16 * hi), c);
  }
  #pragma unroll
  for (int r = 0; r < 8; ++r)
    attnOut[(size_t)(tm * 16 + r + 8 * hi) * DM_ + ncol] = c[r];
}

// ---------- kernel 4: gate + residual + LayerNorm ----------
__global__ void epilogue_kernel(const float* __restrict__ x,
                                const float* __restrict__ attnOut,
                                const float* __restrict__ acc,
                                const float* __restrict__ cth,
                                const float* __restrict__ cscale,
                                const float* __restrict__ gamma,
                                const float* __restrict__ beta,
                                float* __restrict__ out) {
  __shared__ float red1[8], red2[8];
  const int row = blockIdx.x;
  const int tid = threadIdx.x;
  const int wv = tid >> 5, lane = tid & 31;

  const float ncausal = 0.5f * (float)T_ * (float)(T_ + 1);
  const float mean_e  = acc[0] / (ncausal * (float)(B_ * H_) + 1e-9f);
  const float gate    = 1.0f / (1.0f + __expf(-10.0f * (mean_e - cth[0])));
  const float sc      = cscale[0] * gate;

  const float* xr = x + (size_t)row * DM_;
  const float* ar = attnOut + (size_t)row * DM_;

  float y[4];
  float s = 0.f;
  #pragma unroll
  for (int i = 0; i < 4; ++i) {
    const int c = tid + i * 256;
    y[i] = xr[c] + sc * ar[c];
    s += y[i];
  }
  s = wave_sum(s);
  if (lane == 0) red1[wv] = s;
  __syncthreads();
  float mu = 0.f;
  #pragma unroll
  for (int i = 0; i < 8; ++i) mu += red1[i];
  mu *= (1.0f / DM_);

  float v = 0.f;
  #pragma unroll
  for (int i = 0; i < 4; ++i) { const float d = y[i] - mu; v += d * d; }
  v = wave_sum(v);
  if (lane == 0) red2[wv] = v;
  __syncthreads();
  float var = 0.f;
  #pragma unroll
  for (int i = 0; i < 8; ++i) var += red2[i];
  var *= (1.0f / DM_);
  const float inv = rsqrtf(var + LN_EPS_);

  float* outr = out + (size_t)row * DM_;
  #pragma unroll
  for (int i = 0; i < 4; ++i) {
    const int c = tid + i * 256;
    outr[c] = (y[i] - mu) * inv * gamma[c] + beta[c];
  }
}

// ---------- launch ----------
extern "C" void kernel_launch(void* const* d_in, const int* in_sizes, int n_in,
                              void* d_out, int out_size, void* d_ws, size_t ws_size,
                              hipStream_t stream) {
  const float* x      = (const float*)d_in[0];
  const float* Wq     = (const float*)d_in[1];
  const float* Wk     = (const float*)d_in[2];
  const float* Wv     = (const float*)d_in[3];
  const float* Wo     = (const float*)d_in[4];
  const float* cth    = (const float*)d_in[5];
  const float* cscale = (const float*)d_in[6];
  const float* gamma  = (const float*)d_in[7];
  const float* beta   = (const float*)d_in[8];
  float* out = (float*)d_out;

  char* ws = (char*)d_ws;
  float* acc = (float*)ws;
  size_t off = 256;
  const size_t xBytes = (size_t)B_ * T_ * DM_ * sizeof(unsigned short);   // 8 MB
  const size_t wBytes = (size_t)DM_ * DM_ * sizeof(unsigned short);       // 2 MB
  unsigned short* xbf = (unsigned short*)(ws + off); off += xBytes;
  unsigned short* WqT = (unsigned short*)(ws + off); off += wBytes;
  unsigned short* WkT = (unsigned short*)(ws + off); off += wBytes;
  unsigned short* WvT = (unsigned short*)(ws + off); off += wBytes;
  unsigned short* WoT = (unsigned short*)(ws + off); off += wBytes;
  unsigned short* Qbf = (unsigned short*)(ws + off); off += xBytes;
  unsigned short* Kbf = (unsigned short*)(ws + off); off += xBytes;
  unsigned short* Vt  = (unsigned short*)(ws + off); off += xBytes;
  unsigned short* Obf = (unsigned short*)(ws + off); off += xBytes;
  float* attnOut = (float*)(ws + off);

  hipMemsetAsync(acc, 0, sizeof(float), stream);   // graph-capturable

  {  // x -> bf16
    const int n = B_ * T_ * DM_;
    cvt_bf16_kernel<<<n / 8 / 256, 256, 0, stream>>>(x, xbf, n);
  }
  {  // weights -> bf16 transposed
    dim3 grid(DM_ / 32, DM_ / 32, 4), block(32, 8);
    transpose_w_kernel<<<grid, block, 0, stream>>>(Wq, Wk, Wv, Wo,
                                                   WqT, WkT, WvT, WoT);
  }
  {
    const int waves = 3 * (B_ * T_ / 16) * (DM_ / 16);
    qkv_proj_kernel<<<waves / 8, 256, 0, stream>>>(xbf, WqT, WkT, WvT,
                                                   Qbf, Kbf, Vt);
  }
  {
    const int blocks = B_ * H_ * (T_ / 16);
    const size_t lds = (size_t)16 * T_ * 4 + (size_t)16 * T_ * 2 + 128;
    attention_kernel<<<blocks, 256, lds, stream>>>(Qbf, Kbf, Vt, Obf, acc);
  }
  {
    const int waves = (B_ * T_ / 16) * (DM_ / 16);
    out_proj_kernel<<<waves / 8, 256, 0, stream>>>(Obf, WoT, attnOut);
  }
  epilogue_kernel<<<B_ * T_, 256, 0, stream>>>(x, attnOut, acc, cth, cscale,
                                               gamma, beta, out);
}
